// MambaBlock_62062277427577
// MI455X (gfx1250) — matrix operand order repaired and checked by reading, not compiled, not run on hardware
//
#include <hip/hip_runtime.h>
#include <math.h>

// ---------------- problem constants (match reference) ----------------
#define D_MODEL 1024
#define STATE_N 16
#define BATCH   4
#define SEQ_L   4096
#define MROWS   (BATCH * SEQ_L)   // 16384
#define LN_EPS  1e-5f

// ---------------- CDNA5 types ----------------
typedef __attribute__((ext_vector_type(16))) __bf16 v16bf;
typedef __attribute__((ext_vector_type(8)))  float  v8f;
typedef __attribute__((ext_vector_type(4)))  unsigned v4u;
typedef __attribute__((ext_vector_type(8)))  int v8i;
typedef __attribute__((ext_vector_type(4)))  int v4i;

union Frag {
    v16bf v;
    unsigned u[8];
};

// fp32 -> bf16 RNE helpers
__device__ __forceinline__ unsigned short f2bf(float a) {
    unsigned ua = __builtin_bit_cast(unsigned, a);
    unsigned ra = ((ua >> 16) & 1u) + 0x7FFFu;
    return (unsigned short)((ua + ra) >> 16);
}
__device__ __forceinline__ unsigned pack2(float a, float b) {
    unsigned ua = __builtin_bit_cast(unsigned, a);
    unsigned ub = __builtin_bit_cast(unsigned, b);
    unsigned ra = ((ua >> 16) & 1u) + 0x7FFFu;
    unsigned rb = ((ub >> 16) & 1u) + 0x7FFFu;
    return (((ua + ra) >> 16) & 0xFFFFu) | ((ub + rb) & 0xFFFF0000u);
}
__device__ __forceinline__ float bf2f(unsigned short s) {
    return __builtin_bit_cast(float, ((unsigned)s) << 16);
}

// ---------------- WMMA fragment loaders (ISA 7.12.2 layouts) ----------------
// LDS tiles are row-major, 32 bf16 (16 dwords) per row; TDM writes them that way.
__device__ __forceinline__ void load_a_frag(const unsigned* __restrict__ Ls,
                                            int rowBase, int lane, Frag& f) {
    int row = rowBase + (lane & 15);
    int kb2 = (lane >> 4) ? 4 : 0;          // K base 0 or 8 (in dwords /2)
    const unsigned* p = Ls + row * 16 + kb2;
#pragma unroll
    for (int i = 0; i < 4; ++i) f.u[i] = p[i];
#pragma unroll
    for (int i = 0; i < 4; ++i) f.u[4 + i] = p[8 + i];
}
__device__ __forceinline__ void load_b_frag(const unsigned* __restrict__ Ls,
                                            int colBase, int lane, Frag& f) {
    int n = colBase + (lane & 15);
    int kb2 = (lane >> 4) ? 4 : 0;
    const unsigned* p = Ls + n * 16 + kb2;
#pragma unroll
    for (int i = 0; i < 4; ++i) f.u[i] = p[i];
#pragma unroll
    for (int i = 0; i < 4; ++i) f.u[4 + i] = p[8 + i];
}
__device__ __forceinline__ v8f wmma_bf16(const Frag& a, const Frag& b, v8f c) {
    return __builtin_amdgcn_wmma_f32_16x16x32_bf16(false, a.v, false, b.v,
                                                   (short)0, c, false, false);
}

// ---------------- Tensor Data Mover: 2D bf16 tile -> LDS ----------------
// D# built per CDNA5 ISA. Tile treated as its own tensor
// (tensor dims == tile dims, so no OOB path), row stride = K elements.
// LDS result: [tile_h][tile_w] contiguous row-major == fragment layout.
// This toolchain exposes the 6-arg builtin (extra int32x8 group, zero for 2D).
__device__ __forceinline__ unsigned lds_off(const void* p) {
    return (unsigned)(unsigned long long)p;   // flat LDS aperture: offset = addr[31:0]
}
__device__ __forceinline__ void tdm_load_tile_bf16(unsigned ldsOff,
                                                   const unsigned short* g,
                                                   unsigned tile_w, unsigned tile_h,
                                                   unsigned row_stride_elems) {
    unsigned long long ga = (unsigned long long)g;
    v4u g0;
    g0[0] = 1u;                                            // count=1 (valid user D#)
    g0[1] = ldsOff;                                        // lds_addr
    g0[2] = (unsigned)(ga & 0xFFFFFFFFu);                  // global_addr[31:0]
    g0[3] = (unsigned)((ga >> 32) & 0x01FFFFFFu) | (2u << 30);  // addr[56:32] | type=2
    v8i g1;
    g1[0] = (int)(1u << 16);                               // data_size=1 (2 bytes)
    g1[1] = (int)((tile_w & 0xFFFFu) << 16);               // tensor_dim0 low16 @63:48
    g1[2] = (int)((tile_w >> 16) & 0xFFFFu)                // tensor_dim0 hi @79:64
          | (int)((tile_h & 0xFFFFu) << 16);               // tensor_dim1 low16 @95:80
    g1[3] = (int)((tile_h >> 16) & 0xFFFFu)                // tensor_dim1 hi @111:96
          | (int)(tile_w << 16);                           // tile_dim0 @127:112
    g1[4] = (int)(tile_h & 0xFFFFu);                       // tile_dim1 @143:128 (dim2=0)
    g1[5] = (int)row_stride_elems;                         // tensor_dim0_stride low32 @191:160
    g1[6] = 0;                                             // stride hi16 / dim1_stride low
    g1[7] = 0;
    v4i g2 = {0, 0, 0, 0};
    v4i g3 = {0, 0, 0, 0};
    v8i g4 = {0, 0, 0, 0, 0, 0, 0, 0};
    __builtin_amdgcn_tensor_load_to_lds(g0, g1, g2, g3, g4, 0);
}

// =====================================================================
// Kernel 0: bulk fp32 -> packed bf16 conversion (pairs)
// =====================================================================
__global__ __launch_bounds__(256)
void f32_to_bf16_pairs(const float* __restrict__ in, unsigned* __restrict__ out,
                       int npairs) {
    int i = blockIdx.x * 256 + threadIdx.x;
    if (i < npairs) {
        float2 v = ((const float2*)in)[i];
        out[i] = pack2(v.x, v.y);
    }
}

// =====================================================================
// Kernel 1: fused in_proj (dual WMMA GEMM, TDM-fed, double buffered).
// u = X*Wi^T + bi ; z = X*Wz^T + bz
// Epilogue: Ub (bf16), delta = softplus(u+dbias), sg = silu(z).
// Block tile 128x128, BK=32, 256 threads (8 waves, 4x2 wave grid).
// =====================================================================
__global__ __launch_bounds__(256)
void gemm_inproj_dual(const unsigned short* __restrict__ Xb,
                      const unsigned short* __restrict__ Wib, const float* __restrict__ bi,
                      const unsigned short* __restrict__ Wzb, const float* __restrict__ bz,
                      const float* __restrict__ dbias,
                      unsigned short* __restrict__ Ub, float* __restrict__ Delta,
                      float* __restrict__ SG) {
    __shared__ unsigned As[2][128 * 16];
    __shared__ unsigned Bu[2][128 * 16];
    __shared__ unsigned Bz[2][128 * 16];

    const int K = D_MODEL, N = D_MODEL;
    const int tid  = threadIdx.x;
    const int lane = tid & 31;
    const int wave = tid >> 5;
    const int waveM = wave >> 1;
    const int waveN = wave & 1;
    const int m0 = blockIdx.y * 128;
    const int n0 = blockIdx.x * 128;
    const bool issuer = (wave == 0);

    v8f accU[2][4] = {};
    v8f accZ[2][4] = {};

    if (issuer) {
        tdm_load_tile_bf16(lds_off(As[0]), Xb  + (size_t)m0 * K, 32, 128, K);
        tdm_load_tile_bf16(lds_off(Bu[0]), Wib + (size_t)n0 * K, 32, 128, K);
        tdm_load_tile_bf16(lds_off(Bz[0]), Wzb + (size_t)n0 * K, 32, 128, K);
    }
    for (int it = 0; it < K / 32; ++it) {
        const int buf = it & 1;
        if (issuer) __builtin_amdgcn_s_wait_tensorcnt(0);
        __syncthreads();                         // tile it ready; compute it-1 done
        if (issuer && it + 1 < K / 32) {
            const int kt = (it + 1) * 32;
            tdm_load_tile_bf16(lds_off(As[buf ^ 1]), Xb  + (size_t)m0 * K + kt, 32, 128, K);
            tdm_load_tile_bf16(lds_off(Bu[buf ^ 1]), Wib + (size_t)n0 * K + kt, 32, 128, K);
            tdm_load_tile_bf16(lds_off(Bz[buf ^ 1]), Wzb + (size_t)n0 * K + kt, 32, 128, K);
        }
        Frag a[2], fu[4], fz[4];
        load_a_frag(As[buf], waveM * 32,      lane, a[0]);
        load_a_frag(As[buf], waveM * 32 + 16, lane, a[1]);
#pragma unroll
        for (int j = 0; j < 4; ++j) {
            load_b_frag(Bu[buf], waveN * 64 + 16 * j, lane, fu[j]);
            load_b_frag(Bz[buf], waveN * 64 + 16 * j, lane, fz[j]);
        }
#pragma unroll
        for (int mi = 0; mi < 2; ++mi)
#pragma unroll
            for (int ni = 0; ni < 4; ++ni) {
                accU[mi][ni] = wmma_bf16(a[mi], fu[ni], accU[mi][ni]);
                accZ[mi][ni] = wmma_bf16(a[mi], fz[ni], accZ[mi][ni]);
            }
    }

    const int colLane = lane & 15;
    const int rowHi   = (lane >> 4) * 8;
#pragma unroll
    for (int mi = 0; mi < 2; ++mi) {
#pragma unroll
        for (int ni = 0; ni < 4; ++ni) {
            int col = n0 + waveN * 64 + ni * 16 + colLane;
            float biv = bi[col], bzv = bz[col], dbv = dbias[col];
#pragma unroll
            for (int r = 0; r < 8; ++r) {
                int m = m0 + waveM * 32 + mi * 16 + r + rowHi;
                size_t idx = (size_t)m * N + col;
                float uv = accU[mi][ni][r] + biv;
                Ub[idx] = f2bf(uv);
                float t = uv + dbv;
                Delta[idx] = (t > 20.f) ? t : log1pf(__expf(t));
                float zv = accZ[mi][ni][r] + bzv;
                SG[idx] = zv / (1.f + __expf(-zv));     // silu(z)
            }
        }
    }
}

// =====================================================================
// Kernel 2: Bp = U*WB^T + bB ; Cp = U*WC^T + bC  (N=16: one WMMA tile wide)
// Block tile 256x16, TDM-fed, double buffered.
// =====================================================================
__global__ __launch_bounds__(256)
void gemm_bc_dual(const unsigned short* __restrict__ Ub,
                  const unsigned short* __restrict__ WBb, const float* __restrict__ bB,
                  const unsigned short* __restrict__ WCb, const float* __restrict__ bC,
                  float* __restrict__ Bp, float* __restrict__ Cp) {
    __shared__ unsigned As[2][256 * 16];
    __shared__ unsigned Wb[2][16 * 16];
    __shared__ unsigned Wc[2][16 * 16];

    const int K = D_MODEL;
    const int tid  = threadIdx.x;
    const int lane = tid & 31;
    const int wave = tid >> 5;
    const int m0 = blockIdx.x * 256;
    const bool issuer = (wave == 0);

    v8f accB[2] = {};
    v8f accC[2] = {};

    if (issuer) {
        tdm_load_tile_bf16(lds_off(As[0]), Ub  + (size_t)m0 * K, 32, 256, K);
        tdm_load_tile_bf16(lds_off(Wb[0]), WBb, 32, 16, K);
        tdm_load_tile_bf16(lds_off(Wc[0]), WCb, 32, 16, K);
    }
    for (int it = 0; it < K / 32; ++it) {
        const int buf = it & 1;
        if (issuer) __builtin_amdgcn_s_wait_tensorcnt(0);
        __syncthreads();
        if (issuer && it + 1 < K / 32) {
            const int kt = (it + 1) * 32;
            tdm_load_tile_bf16(lds_off(As[buf ^ 1]), Ub  + (size_t)m0 * K + kt, 32, 256, K);
            tdm_load_tile_bf16(lds_off(Wb[buf ^ 1]), WBb + kt, 32, 16, K);
            tdm_load_tile_bf16(lds_off(Wc[buf ^ 1]), WCb + kt, 32, 16, K);
        }
        Frag a0, a1, fb, fc;
        load_a_frag(As[buf], wave * 32,      lane, a0);
        load_a_frag(As[buf], wave * 32 + 16, lane, a1);
        load_b_frag(Wb[buf], 0, lane, fb);
        load_b_frag(Wc[buf], 0, lane, fc);
        accB[0] = wmma_bf16(a0, fb, accB[0]);
        accB[1] = wmma_bf16(a1, fb, accB[1]);
        accC[0] = wmma_bf16(a0, fc, accC[0]);
        accC[1] = wmma_bf16(a1, fc, accC[1]);
    }

    const int col   = lane & 15;
    const int rowHi = (lane >> 4) * 8;
    float bBv = bB[col], bCv = bC[col];
#pragma unroll
    for (int mi = 0; mi < 2; ++mi)
#pragma unroll
        for (int r = 0; r < 8; ++r) {
            int m = m0 + wave * 32 + mi * 16 + r + rowHi;
            Bp[(size_t)m * STATE_N + col] = accB[mi][r] + bBv;
            Cp[(size_t)m * STATE_N + col] = accC[mi][r] + bCv;
        }
}

// =====================================================================
// Kernel 3: selective scan + silu gating (fp32 VALU; sequential in L).
// lane = n + 16*dsub; 16-lane shuffle butterfly reduces sum_n h*C.
// Emits g = y*silu(z) directly in bf16 (out_proj A operand).
// =====================================================================
__global__ __launch_bounds__(256)
void selective_scan(const unsigned short* __restrict__ Ub, const float* __restrict__ Delta,
                    const float* __restrict__ Bp, const float* __restrict__ Cp,
                    const float* __restrict__ A,  const float* __restrict__ Dp,
                    const float* __restrict__ SG, unsigned short* __restrict__ Gb) {
    const int lane = threadIdx.x & 31;
    const int n    = lane & 15;
    const int dsub = lane >> 4;
    const int wave = threadIdx.x >> 5;
    const int b  = blockIdx.x / (D_MODEL / 16);
    const int db = blockIdx.x % (D_MODEL / 16);
    const int d  = db * 16 + wave * 2 + dsub;

    const float a  = A[d * STATE_N + n];
    const float dp = Dp[d];
    float h = 0.f;

    for (int l = 0; l < SEQ_L; ++l) {
        size_t row  = (size_t)b * SEQ_L + l;
        size_t idxD = row * D_MODEL + d;
        float uv = bf2f(Ub[idxD]);
        float dt = Delta[idxD];
        float bn = Bp[row * STATE_N + n];
        float cn = Cp[row * STATE_N + n];
        h = __expf(dt * a) * h + (dt * uv) * bn;
        float p = h * cn;
#pragma unroll
        for (int off = 8; off > 0; off >>= 1)
            p += __shfl_xor(p, off, 32);       // stays inside the 16-lane group
        float g = (p + dp * uv) * SG[idxD];
        if (n == 0) Gb[idxD] = f2bf(g);
    }
}

// =====================================================================
// Kernel 4: out_proj.  YO = G*Wo^T + bo.  TDM-fed, double buffered.
// =====================================================================
__global__ __launch_bounds__(256)
void gemm_outproj(const unsigned short* __restrict__ Gb,
                  const unsigned short* __restrict__ Wob, const float* __restrict__ bo,
                  float* __restrict__ YO) {
    __shared__ unsigned As[2][128 * 16];
    __shared__ unsigned Bs[2][128 * 16];

    const int K = D_MODEL, N = D_MODEL;
    const int tid  = threadIdx.x;
    const int lane = tid & 31;
    const int wave = tid >> 5;
    const int waveM = wave >> 1;
    const int waveN = wave & 1;
    const int m0 = blockIdx.y * 128;
    const int n0 = blockIdx.x * 128;
    const bool issuer = (wave == 0);

    v8f acc[2][4] = {};

    if (issuer) {
        tdm_load_tile_bf16(lds_off(As[0]), Gb  + (size_t)m0 * K, 32, 128, K);
        tdm_load_tile_bf16(lds_off(Bs[0]), Wob + (size_t)n0 * K, 32, 128, K);
    }
    for (int it = 0; it < K / 32; ++it) {
        const int buf = it & 1;
        if (issuer) __builtin_amdgcn_s_wait_tensorcnt(0);
        __syncthreads();
        if (issuer && it + 1 < K / 32) {
            const int kt = (it + 1) * 32;
            tdm_load_tile_bf16(lds_off(As[buf ^ 1]), Gb  + (size_t)m0 * K + kt, 32, 128, K);
            tdm_load_tile_bf16(lds_off(Bs[buf ^ 1]), Wob + (size_t)n0 * K + kt, 32, 128, K);
        }
        Frag a[2], fb[4];
        load_a_frag(As[buf], waveM * 32,      lane, a[0]);
        load_a_frag(As[buf], waveM * 32 + 16, lane, a[1]);
#pragma unroll
        for (int j = 0; j < 4; ++j)
            load_b_frag(Bs[buf], waveN * 64 + 16 * j, lane, fb[j]);
#pragma unroll
        for (int mi = 0; mi < 2; ++mi)
#pragma unroll
            for (int ni = 0; ni < 4; ++ni)
                acc[mi][ni] = wmma_bf16(a[mi], fb[ni], acc[mi][ni]);
    }

    const int colLane = lane & 15;
    const int rowHi   = (lane >> 4) * 8;
#pragma unroll
    for (int mi = 0; mi < 2; ++mi)
#pragma unroll
        for (int ni = 0; ni < 4; ++ni) {
            int col = n0 + waveN * 64 + ni * 16 + colLane;
            float bov = bo[col];
#pragma unroll
            for (int r = 0; r < 8; ++r) {
                int m = m0 + waveM * 32 + mi * 16 + r + rowHi;
                YO[(size_t)m * N + col] = acc[mi][ni][r] + bov;
            }
        }
}

// =====================================================================
// Kernel 5: out = x + layernorm(yo)*gamma + beta  (one block per row)
// =====================================================================
__global__ __launch_bounds__(256)
void ln_residual(const float* __restrict__ X, const float* __restrict__ YO,
                 const float* __restrict__ gamma, const float* __restrict__ beta,
                 float* __restrict__ out) {
    __shared__ float r1[256];
    __shared__ float r2[256];
    const int row = blockIdx.x;
    const int tid = threadIdx.x;
    const size_t base = (size_t)row * D_MODEL;

    float s = 0.f, q = 0.f;
    for (int c = tid; c < D_MODEL; c += 256) {
        float v = YO[base + c];
        s += v; q += v * v;
    }
    r1[tid] = s; r2[tid] = q;
    __syncthreads();
    for (int off = 128; off > 0; off >>= 1) {
        if (tid < off) { r1[tid] += r1[tid + off]; r2[tid] += r2[tid + off]; }
        __syncthreads();
    }
    float mu  = r1[0] * (1.f / D_MODEL);
    float var = r2[0] * (1.f / D_MODEL) - mu * mu;
    float rs  = rsqrtf(var + LN_EPS);
    for (int c = tid; c < D_MODEL; c += 256)
        out[base + c] = X[base + c] + (YO[base + c] - mu) * rs * gamma[c] + beta[c];
}

// =====================================================================
extern "C" void kernel_launch(void* const* d_in, const int* in_sizes, int n_in,
                              void* d_out, int out_size, void* d_ws, size_t ws_size,
                              hipStream_t stream) {
    const float* x     = (const float*)d_in[0];
    const float* Wi    = (const float*)d_in[1];
    const float* bi    = (const float*)d_in[2];
    const float* Wz    = (const float*)d_in[3];
    const float* bz    = (const float*)d_in[4];
    const float* A     = (const float*)d_in[5];
    const float* Dp    = (const float*)d_in[6];
    const float* dbias = (const float*)d_in[7];
    const float* WB    = (const float*)d_in[8];
    const float* bB    = (const float*)d_in[9];
    const float* WC    = (const float*)d_in[10];
    const float* bC    = (const float*)d_in[11];
    const float* Wo    = (const float*)d_in[12];
    const float* bo    = (const float*)d_in[13];
    const float* gamma = (const float*)d_in[14];
    const float* beta  = (const float*)d_in[15];
    float* out = (float*)d_out;

    // ---- workspace layout (256B aligned regions) ----
    const size_t big  = (size_t)MROWS * D_MODEL;     // 16M elements
    const size_t wsz  = (size_t)D_MODEL * D_MODEL;   // 1M elements
    const size_t nsz  = (size_t)STATE_N * D_MODEL;   // 16K elements
    char* p = (char*)d_ws;
    size_t off = 0;
    auto take = [&](size_t bytes) { void* q = p + off; off += (bytes + 255) & ~(size_t)255; return q; };
    unsigned short* Xb  = (unsigned short*)take(big * 2);
    unsigned short* Wib = (unsigned short*)take(wsz * 2);
    unsigned short* Wzb = (unsigned short*)take(wsz * 2);
    unsigned short* Wob = (unsigned short*)take(wsz * 2);
    unsigned short* WBb = (unsigned short*)take(nsz * 2);
    unsigned short* WCb = (unsigned short*)take(nsz * 2);
    unsigned short* Ub  = (unsigned short*)take(big * 2);
    float* Delta = (float*)take(big * 4);            // YO aliases after scan
    float* SG    = (float*)take(big * 4);
    float* Bp    = (float*)take((size_t)MROWS * STATE_N * 4);
    float* Cp    = (float*)take((size_t)MROWS * STATE_N * 4);
    unsigned short* Gb = (unsigned short*)take(big * 2);
    float* YO = Delta;                               // Delta dead after scan

    dim3 blk(256);
    // 0) fp32 -> bf16 conversions
    f32_to_bf16_pairs<<<dim3((unsigned)(big / 2 / 256)), blk, 0, stream>>>(x,  (unsigned*)Xb,  (int)(big / 2));
    f32_to_bf16_pairs<<<dim3((unsigned)(wsz / 2 / 256)), blk, 0, stream>>>(Wi, (unsigned*)Wib, (int)(wsz / 2));
    f32_to_bf16_pairs<<<dim3((unsigned)(wsz / 2 / 256)), blk, 0, stream>>>(Wz, (unsigned*)Wzb, (int)(wsz / 2));
    f32_to_bf16_pairs<<<dim3((unsigned)(wsz / 2 / 256)), blk, 0, stream>>>(Wo, (unsigned*)Wob, (int)(wsz / 2));
    f32_to_bf16_pairs<<<dim3((unsigned)(nsz / 2 / 256)), blk, 0, stream>>>(WB, (unsigned*)WBb, (int)(nsz / 2));
    f32_to_bf16_pairs<<<dim3((unsigned)(nsz / 2 / 256)), blk, 0, stream>>>(WC, (unsigned*)WCb, (int)(nsz / 2));

    dim3 grid1(D_MODEL / 128, MROWS / 128);          // 8 x 128
    gemm_inproj_dual<<<grid1, blk, 0, stream>>>(Xb, Wib, bi, Wzb, bz, dbias, Ub, Delta, SG);

    gemm_bc_dual<<<dim3(MROWS / 256), blk, 0, stream>>>(Ub, WBb, bB, WCb, bC, Bp, Cp);

    selective_scan<<<dim3(BATCH * (D_MODEL / 16)), blk, 0, stream>>>(
        Ub, Delta, Bp, Cp, A, Dp, SG, Gb);

    gemm_outproj<<<grid1, blk, 0, stream>>>(Gb, Wob, bo, YO);

    ln_residual<<<dim3(MROWS), blk, 0, stream>>>(x, YO, gamma, beta, out);
}
